// CNN_CTC_80504866996961
// MI455X (gfx1250) — matrix operand order repaired and checked by reading, not compiled, not run on hardware
//
#include <hip/hip_runtime.h>

// ---------------------------------------------------------------------------
// CTC loss: dense(512->128) + softmax + log(+eps) + CTC forward DP + sum
// B=128, T=1024, D=512, V=128, L=128, S=2L+1=257, BLANK=127
//
// k_gemm: split-bf16 (hi/lo) WMMA v_wmma_f32_16x16x32_bf16, near-fp32 accuracy.
//         Epilogue now writes P = softmax + eps (linear domain).
// k_ctc:  linear-domain scaled forward DP, one wave32 per batch element,
//         alpha in registers (9 states/lane), emission rows double-buffered in
//         LDS via async global->LDS copies (gfx1250 ASYNCcnt path), rescale by
//         wave-wide max every 4 steps, log only at the end.
//
// ws layout (needs ~64.3 MB):
//   [0 , 64MB)           : P      float [B*T][V]   (softmax + eps)
//   [64MB , +128KB)      : W_hi   ushort(bf16), pre-swizzled fragment order
//   [.. , +128KB)        : W_lo   ushort(bf16), same layout
//   [.. , +512B)         : per-batch losses float[128]
// ---------------------------------------------------------------------------

typedef __attribute__((ext_vector_type(16))) __bf16 v16bf;
typedef __attribute__((ext_vector_type(8)))  float  v8f;

#define NB    128
#define NT    1024
#define ND    512
#define NV    128
#define NL    128
#define NS    257          // 2*NL + 1
#define BLANK 127
#define EPSF  (1e-7f)

union Frag16 {
  unsigned int   u[8];
  unsigned short s[16];
  uint4          q[2];
  v16bf          v;
};

static __device__ __forceinline__ unsigned short f32_to_bf16(float f) {
  unsigned int u = __float_as_uint(f);
  u += 0x7FFFu + ((u >> 16) & 1u);          // round-to-nearest-even
  return (unsigned short)(u >> 16);
}
static __device__ __forceinline__ float bf16_to_f32(unsigned short h) {
  return __uint_as_float(((unsigned int)h) << 16);
}

// ---------------------------------------------------------------------------
// Kernel 1: split W (f32 [512][128]) into bf16 hi/lo, pre-swizzled so a GEMM
// lane reads its whole 32x16 B fragment as 16 contiguous bf16 (2x b128).
// ---------------------------------------------------------------------------
__global__ __launch_bounds__(256) void k_prep(const float* __restrict__ W,
                                              unsigned short* __restrict__ whi,
                                              unsigned short* __restrict__ wlo) {
  int idx = blockIdx.x * 256 + threadIdx.x;   // 0 .. 512*128-1
  int k = idx >> 7;                           // D index
  int n = idx & 127;                          // V index
  float f = W[idx];
  unsigned short hi = f32_to_bf16(f);
  unsigned short lo = f32_to_bf16(f - bf16_to_f32(hi));
  int wt = n >> 4, nl = n & 15;
  int kc = k >> 5, kr = k & 31;
  int khalf = (kr >> 3) & 1, kk = (kr & 7) | ((kr >> 4) << 3);
  int lane = (khalf << 4) | nl;
  int dst = (((wt * 16 + kc) * 32) + lane) * 16 + kk;
  whi[dst] = hi;
  wlo[dst] = lo;
}

// ---------------------------------------------------------------------------
// Kernel 2: fused GEMM(+bias) + softmax -> P = softmax + eps, f32 [B*T][V]
// ---------------------------------------------------------------------------
__global__ __launch_bounds__(256) void k_gemm(const float* __restrict__ x,
                                              const float* __restrict__ bias,
                                              const unsigned short* __restrict__ whi,
                                              const unsigned short* __restrict__ wlo,
                                              float* __restrict__ probs) {
  __shared__ unsigned short aHi[32][ND];      // 32 KB
  __shared__ unsigned short aLo[32][ND];      // 32 KB
  __shared__ float tile[32][NV];              // 16 KB

  const int tid = threadIdx.x;
  const long m0 = (long)blockIdx.x * 32;

  const float* xrow = x + (size_t)m0 * ND;
  for (int i = tid; i < 32 * ND; i += 256) {
    float f = xrow[i];
    unsigned short h = f32_to_bf16(f);
    unsigned short l = f32_to_bf16(f - bf16_to_f32(h));
    aHi[i >> 9][i & 511] = h;
    aLo[i >> 9][i & 511] = l;
  }
  __syncthreads();

  const int wave  = tid >> 5;                 // N tile 0..7
  const int lane  = tid & 31;
  const int mrow  = lane & 15;
  const int khalf = lane >> 4;

  v8f acc0 = {0.f,0.f,0.f,0.f,0.f,0.f,0.f,0.f};
  v8f acc1 = {0.f,0.f,0.f,0.f,0.f,0.f,0.f,0.f};

  const unsigned short* paH0 = &aHi[mrow][0];
  const unsigned short* paL0 = &aLo[mrow][0];
  const unsigned short* paH1 = &aHi[mrow + 16][0];
  const unsigned short* paL1 = &aLo[mrow + 16][0];
  const unsigned short* pbH  = &whi[((wave * 16) * 32 + lane) * 16];
  const unsigned short* pbL  = &wlo[((wave * 16) * 32 + lane) * 16];

#pragma unroll
  for (int kc = 0; kc < 16; ++kc) {
    Frag16 bh, bl, ah0, al0, ah1, al1;
    bh.q[0] = ((const uint4*)pbH)[64 * kc];    bh.q[1] = ((const uint4*)pbH)[64 * kc + 1];
    bl.q[0] = ((const uint4*)pbL)[64 * kc];    bl.q[1] = ((const uint4*)pbL)[64 * kc + 1];
    const int kbase = kc * 32 + 8 * khalf;
    ah0.q[0] = *(const uint4*)&paH0[kbase];    ah0.q[1] = *(const uint4*)&paH0[kbase + 16];
    al0.q[0] = *(const uint4*)&paL0[kbase];    al0.q[1] = *(const uint4*)&paL0[kbase + 16];
    ah1.q[0] = *(const uint4*)&paH1[kbase];    ah1.q[1] = *(const uint4*)&paH1[kbase + 16];
    al1.q[0] = *(const uint4*)&paL1[kbase];    al1.q[1] = *(const uint4*)&paL1[kbase + 16];

    acc0 = __builtin_amdgcn_wmma_f32_16x16x32_bf16(false, ah0.v, false, bh.v, (short)0, acc0, false, false);
    acc0 = __builtin_amdgcn_wmma_f32_16x16x32_bf16(false, al0.v, false, bh.v, (short)0, acc0, false, false);
    acc0 = __builtin_amdgcn_wmma_f32_16x16x32_bf16(false, ah0.v, false, bl.v, (short)0, acc0, false, false);
    acc1 = __builtin_amdgcn_wmma_f32_16x16x32_bf16(false, ah1.v, false, bh.v, (short)0, acc1, false, false);
    acc1 = __builtin_amdgcn_wmma_f32_16x16x32_bf16(false, al1.v, false, bh.v, (short)0, acc1, false, false);
    acc1 = __builtin_amdgcn_wmma_f32_16x16x32_bf16(false, ah1.v, false, bl.v, (short)0, acc1, false, false);
  }

#pragma unroll
  for (int j = 0; j < 8; ++j) {
    tile[j + 8 * khalf][wave * 16 + mrow]      = acc0[j];
    tile[16 + j + 8 * khalf][wave * 16 + mrow] = acc1[j];
  }
  __syncthreads();

  // Softmax (+eps): 8 threads per row, 16 cols each, xor-shuffle reduce.
  const int row = tid >> 3;
  const int grp = tid & 7;
  float v[16];
  float mx = -3.4e38f;
#pragma unroll
  for (int j = 0; j < 16; ++j) {
    int n = grp * 16 + j;
    v[j] = tile[row][n] + bias[n];
    mx = fmaxf(mx, v[j]);
  }
#pragma unroll
  for (int m = 1; m < 8; m <<= 1) mx = fmaxf(mx, __shfl_xor(mx, m, 32));
  float s = 0.f;
#pragma unroll
  for (int j = 0; j < 16; ++j) s += expf(v[j] - mx);
#pragma unroll
  for (int m = 1; m < 8; m <<= 1) s += __shfl_xor(s, m, 32);
  const float inv = 1.0f / s;
  float* orow = probs + (size_t)(m0 + row) * NV;
#pragma unroll
  for (int j = 0; j < 16; ++j) {
    int n = grp * 16 + j;
    orow[n] = expf(v[j] - mx) * inv + EPSF;    // softmax + eps
  }
}

// ---------------------------------------------------------------------------
// Kernel 3: CTC forward DP, linear domain with periodic rescaling.
// One wave32 per batch element; alpha in registers (9 states/lane, blocked);
// emission rows double-buffered in LDS via async global->LDS copies.
// ---------------------------------------------------------------------------
__global__ __launch_bounds__(32) void k_ctc(const float* __restrict__ P,
                                            const int* __restrict__ targets,
                                            const int* __restrict__ tlen,
                                            float* __restrict__ loss) {
  __shared__ float rowbuf[2][NV];   // double-buffered emission rows
  __shared__ int   tgt[NL];
  __shared__ float afin[32 * 9];

  const int lane = threadIdx.x;     // wave32
  const int b    = blockIdx.x;
  const float* lp = P + (size_t)b * NT * NV;

  // Stage targets + row 0 (sync path), then async-prefetch row 1 into buf 1.
  for (int i = lane; i < NL; i += 32) tgt[i] = targets[b * NL + i];
  for (int i = lane; i < NV; i += 32) rowbuf[0][i] = lp[i];
  {
    unsigned ldsa = (unsigned)(unsigned long long)&rowbuf[1][0] + (unsigned)(lane * 16);
    unsigned voff = (unsigned)((NV + lane * 4) * 4);
    asm volatile("global_load_async_to_lds_b128 %0, %1, %2"
                 :: "v"(ldsa), "v"(voff), "s"((unsigned long long)lp) : "memory");
  }
  asm volatile("s_wait_dscnt 0" ::: "memory");
  __builtin_amdgcn_wave_barrier();

  // Per-state constants: state s = lane*9 + j.
  int   labi[9];
  float all2[9];
  float vmsk[9];
  float a[9];
#pragma unroll
  for (int j = 0; j < 9; ++j) {
    int s = lane * 9 + j;
    bool valid = (s < NS);
    vmsk[j] = valid ? 1.f : 0.f;
    int lab = BLANK;
    float w2 = 0.f;
    if (valid && (s & 1)) {
      int li = (s - 1) >> 1;
      lab = tgt[li];
      if (s >= 3 && lab != tgt[li - 1]) w2 = 1.f;
    }
    labi[j] = lab;
    all2[j] = w2;
    // t = 0: only s=0 (blank) and s=1 (first label) are live.
    a[j] = (s < 2) ? rowbuf[0][lab] : 0.f;
  }

  float logacc = 0.f;

  for (int t = 1; t < NT; ++t) {
    const int ib = t & 1;
    // Wait for the async prefetch that filled rowbuf[ib].
    asm volatile("s_wait_asynccnt 0" ::: "memory");
    __builtin_amdgcn_wave_barrier();
    // Kick prefetch of row t+1 into the other buffer.
    if (t + 1 < NT) {
      unsigned ldsa = (unsigned)(unsigned long long)&rowbuf[ib ^ 1][0] + (unsigned)(lane * 16);
      unsigned voff = (unsigned)(((t + 1) * NV + lane * 4) * 4);
      asm volatile("global_load_async_to_lds_b128 %0, %1, %2"
                   :: "v"(ldsa), "v"(voff), "s"((unsigned long long)lp) : "memory");
    }

    // Cross-lane neighbors: alpha[lane*9-1], alpha[lane*9-2] from lane-1.
    float nb8 = __shfl_up(a[8], 1);
    float nb7 = __shfl_up(a[7], 1);
    if (lane == 0) { nb8 = 0.f; nb7 = 0.f; }

    float na[9];
#pragma unroll
    for (int j = 0; j < 9; ++j) {
      float p1 = (j == 0) ? nb8 : a[j - 1 < 0 ? 0 : j - 1];
      float p2 = (j == 0) ? nb7 : ((j == 1) ? nb8 : a[j >= 2 ? j - 2 : 0]);
      if (j >= 1) p1 = a[j - 1];
      float e = rowbuf[ib][labi[j]];
      na[j] = vmsk[j] * ((a[j] + p1 + all2[j] * p2) * e);
    }
#pragma unroll
    for (int j = 0; j < 9; ++j) a[j] = na[j];

    // Rescale every 4 steps by the wave-wide max (kept exact via logacc).
    if ((t & 3) == 3) {
      float mx = 0.f;
#pragma unroll
      for (int j = 0; j < 9; ++j) mx = fmaxf(mx, a[j]);
#pragma unroll
      for (int m = 16; m >= 1; m >>= 1) mx = fmaxf(mx, __shfl_xor(mx, m, 32));
      float inv = 1.0f / mx;
#pragma unroll
      for (int j = 0; j < 9; ++j) a[j] *= inv;
      logacc += logf(mx);
    }
  }

  // Gather final alpha, emit per-batch loss.
#pragma unroll
  for (int j = 0; j < 9; ++j) afin[lane * 9 + j] = a[j];
  asm volatile("s_wait_dscnt 0" ::: "memory");
  __builtin_amdgcn_wave_barrier();
  if (lane == 0) {
    int last = 2 * tlen[b];                   // <= 256
    loss[b] = -(logf(afin[last] + afin[last - 1]) + logacc);
  }
}

// ---------------------------------------------------------------------------
// Kernel 4: deterministic sum of 128 per-batch losses -> d_out[0]
// ---------------------------------------------------------------------------
__global__ __launch_bounds__(128) void k_reduce(const float* __restrict__ loss,
                                                float* __restrict__ out) {
  __shared__ float sm[128];
  const int tid = threadIdx.x;
  sm[tid] = loss[tid];
  __syncthreads();
  for (int off = 64; off > 0; off >>= 1) {
    if (tid < off) sm[tid] += sm[tid + off];
    __syncthreads();
  }
  if (tid == 0) out[0] = sm[0];
}

// ---------------------------------------------------------------------------
extern "C" void kernel_launch(void* const* d_in, const int* in_sizes, int n_in,
                              void* d_out, int out_size, void* d_ws, size_t ws_size,
                              hipStream_t stream) {
  const float* x       = (const float*)d_in[0];   // [128,1024,512]
  const float* W       = (const float*)d_in[1];   // [512,128]
  const float* bias    = (const float*)d_in[2];   // [128]
  const int*   targets = (const int*)d_in[3];     // [128,128]
  const int*   tlen    = (const int*)d_in[4];     // [128]
  float*       out     = (float*)d_out;

  char* ws = (char*)d_ws;
  float*          probs = (float*)ws;                                    // 64 MB
  unsigned short* whi   = (unsigned short*)(ws + (size_t)67108864);      // 128 KB
  unsigned short* wlo   = whi + (size_t)ND * NV;                         // 128 KB
  float*          loss  = (float*)(ws + (size_t)67108864 + 262144);      // 512 B

  k_prep  <<<(ND * NV) / 256, 256, 0, stream>>>(W, whi, wlo);
  k_gemm  <<<(NB * NT) / 32, 256, 0, stream>>>(x, bias, whi, wlo, probs);
  k_ctc   <<<NB, 32, 0, stream>>>(probs, targets, tlen, loss);
  k_reduce<<<1, 128, 0, stream>>>(loss, out);
}
